// AttentionModel_57260503990445
// MI455X (gfx1250) — matrix operand order repaired and checked by reading, not compiled, not run on hardware
//
#include <hip/hip_runtime.h>
#include <hip/hip_bf16.h>

// ---------------------------------------------------------------------------
// CDNA5 (gfx1250) attention block, bf16 WMMA + TDM + async global->LDS.
//   - All GEMM B operands pre-transposed to [N][K] bf16, N padded to 128.
//   - GEMM: 256x128x32 block tile, 8 waves (wave32) 4x2, wave tile 64x64
//     (4x4 WMMA fragments), double-buffered LDS.
//   - A tile (256x32) staged by the Tensor Data Mover (tensor_load_to_lds,
//     TENSORcnt); B tile staged with global_load_async_to_lds_b128 (ASYNCcnt).
//   - QKV fused into one GEMM (N = 3*768 = 2304) sharing the x operand.
// ---------------------------------------------------------------------------

typedef __attribute__((ext_vector_type(16))) __bf16 v16bf;
typedef __attribute__((ext_vector_type(8)))  __bf16 v8bf;
typedef __attribute__((ext_vector_type(8)))  float  v8f;
typedef __attribute__((ext_vector_type(4)))  unsigned int u32x4;
typedef __attribute__((ext_vector_type(8)))  unsigned int u32x8;

static __device__ __forceinline__ v16bf cat16(v8bf lo, v8bf hi) {
    return __builtin_shufflevector(lo, hi, 0,1,2,3,4,5,6,7,8,9,10,11,12,13,14,15);
}

// Async DMA of 16 bytes global -> LDS (ASYNCcnt).  VDST = LDS byte offset.
static __device__ __forceinline__ void async_b128(void* lds_ptr, const void* gptr) {
    unsigned int       lds_off = (unsigned int)(unsigned long long)(uintptr_t)lds_ptr;
    unsigned long long gaddr   = (unsigned long long)(uintptr_t)gptr;
    asm volatile("global_load_async_to_lds_b128 %0, %1, off"
                 :: "v"(lds_off), "v"(gaddr) : "memory");
}

static __device__ __forceinline__ void wait_async0() {
#if __has_builtin(__builtin_amdgcn_s_wait_asynccnt)
    __builtin_amdgcn_s_wait_asynccnt(0);
#else
    asm volatile("s_wait_asynccnt 0x0" ::: "memory");
#endif
}

static __device__ __forceinline__ void wait_tensor0() {
#if __has_builtin(__builtin_amdgcn_s_wait_tensorcnt)
    __builtin_amdgcn_s_wait_tensorcnt(0);
#else
    asm volatile("s_wait_tensorcnt 0x0" ::: "memory");
#endif
}

// TDM: load a 2D tile [rows x 32 elems] of bf16 from global (row stride lda
// elements) into LDS at lds_off, rows stored contiguously (64B per row).
// D# per CDNA5 ISA ch.8: group0 = {count/lds_addr/global_addr/type},
// group1 = {data_size, tensor dims, tile dims, dim0 stride}.
static __device__ __forceinline__ void tdm_load_tile_a(unsigned int lds_off,
                                                       unsigned long long gaddr,
                                                       unsigned int lda_elems) {
    u32x4 g0;
    g0[0] = 1u;                                               // count=1 (user D#)
    g0[1] = lds_off;                                          // lds_addr
    g0[2] = (unsigned int)gaddr;                              // global_addr[31:0]
    g0[3] = (unsigned int)((gaddr >> 32) & 0x1FFFFFFull)      // global_addr[56:32]
          | (2u << 30);                                       // type = 2 (image)
    u32x8 g1;
    g1[0] = 1u << 16;            // workgroup_mask=0, data_size=1 (2 bytes)
    g1[1] = 32u << 16;           // tensor_dim0 = 32 (low 16)
    g1[2] = 256u << 16;          // tensor_dim0 hi=0 | tensor_dim1 = 256 (low 16)
    g1[3] = 32u << 16;           // tensor_dim1 hi=0 | tile_dim0 = 32
    g1[4] = 256u;                // tile_dim1 = 256, tile_dim2 = 0
    g1[5] = lda_elems;           // tensor_dim0_stride[31:0] (data_size units)
    g1[6] = 0u;                  // stride hi, dim1_stride lo
    g1[7] = 0u;                  // dim1_stride hi (unused, 2D tile)
    asm volatile("tensor_load_to_lds %0, %1" :: "s"(g0), "s"(g1) : "memory");
}

// fp32 [srcR x srcC] (row stride srcLd) -> bf16 [dstR x dstC], zero padded.
__global__ void cvt_pad_bf16(const float* __restrict__ src, __bf16* __restrict__ dst,
                             int srcR, int srcC, int srcLd, int dstR, int dstC) {
    size_t total  = (size_t)dstR * (size_t)dstC;
    size_t stride = (size_t)gridDim.x * blockDim.x;
    for (size_t i = (size_t)blockIdx.x * blockDim.x + threadIdx.x; i < total; i += stride) {
        size_t r = i / (size_t)dstC;
        size_t c = i % (size_t)dstC;
        float v = 0.0f;
        if (r < (size_t)srcR && c < (size_t)srcC) v = src[r * (size_t)srcLd + c];
        dst[i] = (__bf16)v;
    }
}

// fp32 [R x C] (row stride srcLd) -> bf16 transposed [Cp x Rp], zero padded.
__global__ void transpose_pad_bf16(const float* __restrict__ src, __bf16* __restrict__ dst,
                                   int R, int C, int srcLd, int Rp, int Cp) {
    size_t total  = (size_t)Rp * (size_t)Cp;
    size_t stride = (size_t)gridDim.x * blockDim.x;
    for (size_t i = (size_t)blockIdx.x * blockDim.x + threadIdx.x; i < total; i += stride) {
        size_t c = i / (size_t)Rp;       // output row (source column)
        size_t r = i % (size_t)Rp;       // output col (source row)
        float v = 0.0f;
        if (r < (size_t)R && c < (size_t)C) v = src[r * (size_t)srcLd + c];
        dst[i] = (__bf16)v;
    }
}

// concat three fp32 vectors of length n into dst[3n]
__global__ void pack3_f32(const float* __restrict__ a, const float* __restrict__ b,
                          const float* __restrict__ c, float* __restrict__ dst, int n) {
    int i = blockIdx.x * blockDim.x + threadIdx.x;
    if (i < n)          dst[i] = a[i];
    else if (i < 2 * n) dst[i] = b[i - n];
    else if (i < 3 * n) dst[i] = c[i - 2 * n];
}

// ---------------------------------------------------------------------------
// C[M,N](f32) = A[M,K](bf16 row-major) @ B[N,K](bf16 row-major = B^T) + bias
// Block tile 256x128x32, 8 waves 4x2, wave tile 64x64 (4x4 fragments).
// Requirements: M % 256 == 0, K % 32 == 0, B rows >= ceil(N/128)*128.
// ---------------------------------------------------------------------------
__global__ __launch_bounds__(256)
void gemm_bf16_wmma(const __bf16* __restrict__ A, const __bf16* __restrict__ B,
                    float* __restrict__ C, const float* __restrict__ bias,
                    int M, int N, int K, int lda, int ldb, int ldc) {
    __shared__ __align__(16) __bf16 As[2][256][32];   // 16 KB / buf  (TDM dest)
    __shared__ __align__(16) __bf16 Bs[2][128][32];   // 8 KB / buf  (async dest)

    const int tid   = threadIdx.x;
    const int lane  = tid & 31;
    const int wave  = tid >> 5;        // 0..7
    const int waveM = wave >> 1;       // 0..3  (64-row quarters of 256)
    const int waveN = wave & 1;        // 0..1  (64-col halves of 128)
    const int blockM = blockIdx.y * 256;
    const int blockN = blockIdx.x * 128;

    // A tile base (uniform) for the TDM descriptor; advances 64B per K-step.
    const unsigned long long aTileBase =
        (unsigned long long)(uintptr_t)(A + (size_t)blockM * lda);
    const unsigned int ldsA0 = (unsigned int)(uintptr_t)&As[0][0][0];
    const unsigned int ldsA1 = (unsigned int)(uintptr_t)&As[1][0][0];

    // B staging ownership: 2 chunks of 8 bf16 per thread
    const int bn0  = tid >> 1;                   // B row 0..127
    const int bkc0 = (tid & 1) * 16;             // B k-offset 0 / 16
    const __bf16* bSrc = B + (size_t)(blockN + bn0) * ldb + bkc0;

    const v8f vzero = {0.f, 0.f, 0.f, 0.f, 0.f, 0.f, 0.f, 0.f};
    v8f acc[4][4];
#pragma unroll
    for (int mi = 0; mi < 4; ++mi)
#pragma unroll
        for (int ni = 0; ni < 4; ++ni) acc[mi][ni] = vzero;

    const int T = K >> 5;

    auto issue = [&](int t, int b) {
        if (wave == 0)   // one TDM op per block per step covers the whole A tile
            tdm_load_tile_a(b ? ldsA1 : ldsA0,
                            aTileBase + (unsigned long long)t * 64ull,
                            (unsigned int)lda);
        const __bf16* bp = bSrc + (size_t)t * 32;
        async_b128(&Bs[b][bn0][bkc0 + 0], bp + 0);
        async_b128(&Bs[b][bn0][bkc0 + 8], bp + 8);
    };
    auto waits = [&]() {
        if (wave == 0) wait_tensor0();
        wait_async0();
    };

    // fragment addressing (CDNA5 wave32 layouts)
    const int fr  = lane & 15;
    const int akb = (lane >> 4) * 8;     // A: K = {akb..+7, akb+16..+23}
    const int bkb = (lane >> 4) * 8;     // B: K = {bkb..+7, bkb+16..+23}

    auto compute = [&](int b) {
        v16bf afrag[4];
#pragma unroll
        for (int mi = 0; mi < 4; ++mi) {
            int m = waveM * 64 + mi * 16 + fr;
            afrag[mi] = cat16(*(const v8bf*)&As[b][m][akb],
                              *(const v8bf*)&As[b][m][akb + 16]);
        }
        v16bf bfrag[4];
#pragma unroll
        for (int ni = 0; ni < 4; ++ni) {
            int n = waveN * 64 + ni * 16 + fr;
            bfrag[ni] = cat16(*(const v8bf*)&Bs[b][n][bkb],
                              *(const v8bf*)&Bs[b][n][bkb + 16]);
        }
#pragma unroll
        for (int mi = 0; mi < 4; ++mi)
#pragma unroll
            for (int ni = 0; ni < 4; ++ni)
                acc[mi][ni] = __builtin_amdgcn_wmma_f32_16x16x32_bf16(
                    false, afrag[mi], false, bfrag[ni],
                    (short)0, acc[mi][ni], false, false);
    };

    issue(0, 0);
    int buf = 0;
    for (int t = 0; t < T - 1; ++t) {
        waits();                       // tile t landed (TDM + async)
        __syncthreads();               // published to all waves; old buf free
        issue(t + 1, buf ^ 1);         // overlap next fill with current math
        __builtin_prefetch(bSrc + (size_t)(t + 2) * 32, 0, 1);  // speculative
        compute(buf);
        buf ^= 1;
    }
    waits();
    __syncthreads();
    compute(buf);

    // store: VGPR r -> row = r + 8*(lane>>4), col = lane&15
    const int rbase = (lane >> 4) * 8;
#pragma unroll
    for (int mi = 0; mi < 4; ++mi) {
#pragma unroll
        for (int ni = 0; ni < 4; ++ni) {
            int col = blockN + waveN * 64 + ni * 16 + (lane & 15);
            if (col < N) {
                float bv = bias ? bias[col] : 0.0f;
                int row0 = blockM + waveM * 64 + mi * 16 + rbase;
#pragma unroll
                for (int r = 0; r < 8; ++r)
                    C[(size_t)(row0 + r) * ldc + col] = acc[mi][ni][r] + bv;
            }
        }
    }
}

// ---------------------------------------------------------------------------
// Row-wise softmax: P[row,:] = softmax(scale * S[row,:]) as bf16.
// ---------------------------------------------------------------------------
__global__ __launch_bounds__(256)
void softmax_rows(const float* __restrict__ S, __bf16* __restrict__ P,
                  int cols, float scale) {
    const int row = blockIdx.x;
    const int tid = threadIdx.x;
    const float* s = S + (size_t)row * cols;
    __shared__ float red[256];

    float m = -3.0e38f;
    for (int c = tid; c < cols; c += 256) m = fmaxf(m, s[c] * scale);
    red[tid] = m;
    __syncthreads();
    for (int off = 128; off > 0; off >>= 1) {
        if (tid < off) red[tid] = fmaxf(red[tid], red[tid + off]);
        __syncthreads();
    }
    const float rowmax = red[0];
    __syncthreads();

    float sum = 0.f;
    for (int c = tid; c < cols; c += 256) sum += __expf(s[c] * scale - rowmax);
    red[tid] = sum;
    __syncthreads();
    for (int off = 128; off > 0; off >>= 1) {
        if (tid < off) red[tid] += red[tid + off];
        __syncthreads();
    }
    const float inv = 1.0f / red[0];

    __bf16* p = P + (size_t)row * cols;
    for (int c = tid; c < cols; c += 256)
        p[c] = (__bf16)(__expf(s[c] * scale - rowmax) * inv);
}

// ---------------------------------------------------------------------------
extern "C" void kernel_launch(void* const* d_in, const int* in_sizes, int n_in,
                              void* d_out, int out_size, void* d_ws, size_t ws_size,
                              hipStream_t stream) {
    (void)in_sizes; (void)n_in; (void)out_size; (void)ws_size;

    constexpr int S   = 2048;
    constexpr int V   = 50257;
    constexpr int VP  = 50304;   // V padded to multiple of 128
    constexpr int E   = 768;
    constexpr int E3  = 3 * E;   // fused QKV width = 2304 (18 * 128)

    const float* x  = (const float*)d_in[0];
    const float* Wq = (const float*)d_in[1];
    const float* bq = (const float*)d_in[2];
    const float* Wk = (const float*)d_in[3];
    const float* bk = (const float*)d_in[4];
    const float* Wv = (const float*)d_in[5];
    const float* bv = (const float*)d_in[6];
    const float* Wo = (const float*)d_in[7];
    const float* bo = (const float*)d_in[8];
    float* out = (float*)d_out;

    char* base = (char*)d_ws;
    size_t off = 0;
    auto carve = [&](size_t bytes) -> char* {
        char* p = base + off;
        off = (off + bytes + 255) & ~(size_t)255;
        return p;
    };
    __bf16* xb    = (__bf16*)carve((size_t)S  * VP * 2);  // x [S][VP]
    __bf16* wqkvt = (__bf16*)carve((size_t)E3 * VP * 2);  // [Wq^T;Wk^T;Wv^T] [2304][VP]
    __bf16* wot   = (__bf16*)carve((size_t)VP * E  * 2);  // Wo^T [VP][E]
    float*  bqkv  = (float*) carve((size_t)E3 * 4);       // packed biases
    float*  QKVf  = (float*) carve((size_t)S * E3 * 4);   // fused [S][2304]
    __bf16* Qb    = (__bf16*)carve((size_t)S * E * 2);    // Q [S][E]
    __bf16* Kb    = (__bf16*)carve((size_t)S * E * 2);    // K [S][E]
    __bf16* Vt    = (__bf16*)carve((size_t)E * S * 2);    // V^T [E][S]
    float*  Sc    = (float*) carve((size_t)S * S * 4);
    __bf16* Pb    = (__bf16*)carve((size_t)S * S * 2);
    float*  Cf    = (float*) carve((size_t)S * E * 4);
    __bf16* Cb    = (__bf16*)carve((size_t)S * E * 2);
    float*  Qf = QKVf + 0, *Kf = QKVf + E, *Vf = QKVf + 2 * E;   // ld = E3

    // ---- 1) convert / pad / transpose inputs
    cvt_pad_bf16      <<<1024, 256, 0, stream>>>(x, xb, S, V, V, S, VP);
    transpose_pad_bf16<<<1024, 256, 0, stream>>>(Wq, wqkvt + (size_t)0 * E * VP, V, E, E, VP, E);
    transpose_pad_bf16<<<1024, 256, 0, stream>>>(Wk, wqkvt + (size_t)1 * E * VP, V, E, E, VP, E);
    transpose_pad_bf16<<<1024, 256, 0, stream>>>(Wv, wqkvt + (size_t)2 * E * VP, V, E, E, VP, E);
    transpose_pad_bf16<<<1024, 256, 0, stream>>>(Wo, wot, E, V, V, E, VP);   // [VP][E]
    pack3_f32<<<(E3 + 255) / 256, 256, 0, stream>>>(bq, bk, bv, bqkv, E);

    // ---- 2) fused QKV projection: [S x VP] @ [VP x 2304]
    {
        dim3 grid(E3 / 128, S / 256);
        gemm_bf16_wmma<<<grid, 256, 0, stream>>>(xb, wqkvt, QKVf, bqkv,
                                                 S, E3, VP, VP, VP, E3);
    }
    cvt_pad_bf16      <<<1024, 256, 0, stream>>>(Qf, Qb, S, E, E3, S, E);
    cvt_pad_bf16      <<<1024, 256, 0, stream>>>(Kf, Kb, S, E, E3, S, E);
    transpose_pad_bf16<<<1024, 256, 0, stream>>>(Vf, Vt, S, E, E3, S, E);    // [E][S]

    // ---- 3) scores = Q @ K^T  [S x S], K-dim = E
    {
        dim3 grid(S / 128, S / 256);
        gemm_bf16_wmma<<<grid, 256, 0, stream>>>(Qb, Kb, Sc, nullptr, S, S, E, E, E, S);
    }

    // ---- 4) softmax(scale * scores) -> bf16 probs
    const float scale = 1.0f / sqrtf((float)E);
    softmax_rows<<<S, 256, 0, stream>>>(Sc, Pb, S, scale);

    // ---- 5) ctx = P @ (V^T)^T  [S x E], K-dim = S
    {
        dim3 grid(E / 128, S / 256);
        gemm_bf16_wmma<<<grid, 256, 0, stream>>>(Pb, Vt, Cf, nullptr, S, E, S, S, S, E);
    }
    cvt_pad_bf16<<<1024, 256, 0, stream>>>(Cf, Cb, S, E, E, S, E);

    // ---- 6) out = ctx @ (Wo^T)^T  [S x V], K-dim = E
    {
        dim3 grid(VP / 128, S / 256);
        gemm_bf16_wmma<<<grid, 256, 0, stream>>>(Cb, wot, out, bo, S, V, E, E, E, V);
    }
}
